// Q_network_35218731827714
// MI455X (gfx1250) — matrix operand order repaired and checked by reading, not compiled, not run on hardware
//
#include <hip/hip_runtime.h>
#include <hip/hip_bf16.h>
#include <stdint.h>

// ---------------------------------------------------------------------------
// Q-network GNN forward for MI455X (gfx1250), wave32 + WMMA f16->f32.
// GEMMs run as v_wmma_f32_16x16x32_f16, fp32 accumulate. Per-row-tile fused
// 3-layer MLPs keep the 512->256->256->128 chain in LDS. Weights are
// pre-swizzled once per launch into WMMA-B fragment-native order so every
// B fragment is one aligned 32-byte global load (L2-resident), and every A
// fragment is two contiguous ds_load_b128. Each wave register-blocks 2 M
// row-tiles so each B fragment feeds two WMMAs (~1 b128 pair per 2 WMMA).
// ---------------------------------------------------------------------------

typedef __attribute__((ext_vector_type(16))) _Float16 v16h;
typedef __attribute__((ext_vector_type(8)))  _Float16 v8h;
typedef __attribute__((ext_vector_type(8)))  float    v8f;

#define DEV __device__ __forceinline__

// A matrix 16x32 f16 from row-major LDS tile (stride lda halves).
// Lane layout: row = lane&15; kb = (lane>=16)?8:0.
// elements 0..7  <-> K = kb..kb+7; elements 8..15 <-> K = 16+kb..16+kb+7
// => two aligned 16B LDS loads.
DEV v16h load_a16(const _Float16* p) {
  const v8h lo = *(const v8h*)p;
  const v8h hi = *(const v8h*)(p + 16);
  return __builtin_shufflevector(lo, hi, 0, 1, 2, 3, 4, 5, 6, 7,
                                 8, 9, 10, 11, 12, 13, 14, 15);
}

// ---- fused MLP layer: Out[64,NOUT] = act(A[64,KIN] @ W + b) ---------------
// Block = 256 threads = 8 waves. Wave w: row tiles {2*(w&1), 2*(w&1)+1},
// col tiles [ (w>>1)*CPW , +CPW ). gWs = swizzled f16 weights in
// [K/32][N/16][lane(32)][16] fragment-native order.

enum OutMode { OM_LDSH, OM_LDSF, OM_GST, OM_GADD };

template <int KIN, int NOUT, bool RELU, OutMode OM>
DEV void mlp_layer(const _Float16* lA, const _Float16* __restrict__ gWs,
                   const float* __restrict__ bias, _Float16* lOutH,
                   float* lOutF, float* gOut, int ldg, int row0) {
  const int tid = threadIdx.x, lane = tid & 31, wave = tid >> 5;
  constexpr int CPW = NOUT / 64;              // 16-wide col tiles per wave
  constexpr int NT  = NOUT / 16;              // col tiles total
  const int rts = (wave & 1) * 2;             // first of 2 row tiles
  const int ct0 = (wave >> 1) * CPW;          // first col tile of this wave
  v8f acc[2][CPW];
#pragma unroll
  for (int m = 0; m < 2; ++m)
#pragma unroll
    for (int c = 0; c < CPW; ++c)
#pragma unroll
      for (int i = 0; i < 8; ++i) acc[m][c][i] = 0.f;

  const int arow = lane & 15;
  const int akb  = (lane >> 4) << 3;
  const _Float16* lArt0 = lA + (rts * 16 + arow) * KIN + akb;
  const _Float16* lArt1 = lArt0 + 16 * KIN;

#pragma unroll
  for (int kc = 0; kc < KIN; kc += 32) {
    const v16h a0 = load_a16(lArt0 + kc);
    const v16h a1 = load_a16(lArt1 + kc);
    const _Float16* wb =
        gWs + (size_t)(((kc >> 5) * NT + ct0) * 512 + lane * 16);
    if (kc + 32 < KIN) __builtin_prefetch((const void*)(wb + NT * 512), 0, 1);
#pragma unroll
    for (int c = 0; c < CPW; ++c) {
      const v16h b = *(const v16h*)(wb + c * 512);
      acc[0][c] = __builtin_amdgcn_wmma_f32_16x16x32_f16(
          false, a0, false, b, (short)0, acc[0][c], false, false);
      acc[1][c] = __builtin_amdgcn_wmma_f32_16x16x32_f16(
          false, a1, false, b, (short)0, acc[1][c], false, false);
    }
  }

  // epilogue: bias (+relu), scatter per C layout (VGPR i -> M=i or i+8)
  const int col = lane & 15, rb = (lane >> 4) << 3;
#pragma unroll
  for (int m = 0; m < 2; ++m)
#pragma unroll
    for (int c = 0; c < CPW; ++c) {
      const int cc = (ct0 + c) * 16 + col;
      const float bv = bias[cc];
#pragma unroll
      for (int i = 0; i < 8; ++i) {
        float v = acc[m][c][i] + bv;
        if (RELU) v = fmaxf(v, 0.f);
        const int r = (rts + m) * 16 + rb + i;
        if (OM == OM_LDSH)      lOutH[r * NOUT + cc] = (_Float16)v;
        else if (OM == OM_LDSF) lOutF[r * NOUT + cc] = v;
        else if (OM == OM_GST)  gOut[(size_t)(row0 + r) * ldg + cc] = v;
        else                    gOut[(size_t)(row0 + r) * ldg + cc] += v;
      }
    }
  if (OM == OM_LDSH || OM == OM_LDSF) __syncthreads();
}

// wave32 sum reduction
DEV float wred(float v) {
#pragma unroll
  for (int o = 16; o; o >>= 1) v += __shfl_xor(v, o, 32);
  return v;
}

// pack two halves into one dword store
DEV void st2h(_Float16* p, float a, float b) {
  union { _Float16 h[2]; uint32_t u; } pk;
  pk.h[0] = (_Float16)a; pk.h[1] = (_Float16)b;
  *(uint32_t*)p = pk.u;
}

// ---------------------------------------------------------------------------
// utility kernels
// ---------------------------------------------------------------------------

// swizzle f32 W[K,N] -> f16 fragment-native [K/32][N/16][lane(32)][16]
__global__ void k_wswz(const float* __restrict__ src, _Float16* __restrict__ dst,
                       int K, int N) {
  const int i = blockIdx.x * 256 + threadIdx.x;
  if (i >= K * N) return;
  const int j = i & 15;
  const int l = (i >> 4) & 31;
  const int blk = i >> 9;
  const int nt = N >> 4;
  const int kc = (blk / nt) << 5;
  const int ct = blk % nt;
  const int col = ct * 16 + (l & 15);
  const int k = kc + ((l >> 4) << 3) + (j & 7) + ((j >> 3) << 4);
  dst[i] = (_Float16)src[(size_t)k * N + col];
}
__global__ void k_zero(float* p, int n) {
  int i = blockIdx.x * 256 + threadIdx.x;
  if (i < n) p[i] = 0.f;
}
__global__ void k_bcast_e(const float* __restrict__ e_init, float* __restrict__ e, size_t n) {
  size_t i = (size_t)blockIdx.x * 256 + threadIdx.x;
  if (i < n) e[i] = e_init[i & 127];
}
__global__ void k_axpy(float* __restrict__ d, const float* __restrict__ s, size_t n) {
  size_t i = (size_t)blockIdx.x * 256 + threadIdx.x;
  if (i < n) d[i] += s[i];
}

// edge encoder on the constant one-hot edge_attr (row 0): one tiny block.
__global__ void k_edge_enc(const float* W0, const float* W1, const float* W2,
                           const float* b0, const float* b1, const float* b2,
                           const float* ls, const float* lb, float* e_init) {
  __shared__ float h0[256], h1[256], h2[128], red[2];
  const int t = threadIdx.x;
  h0[t] = fmaxf(W0[t] + b0[t], 0.f);          // one-hot index 0 -> W0 row 0
  __syncthreads();
  { float s = 0.f; for (int k = 0; k < 256; ++k) s += h0[k] * W1[k * 256 + t];
    h1[t] = fmaxf(s + b1[t], 0.f); }
  __syncthreads();
  if (t < 128) { float s = 0.f; for (int k = 0; k < 256; ++k) s += h1[k] * W2[k * 128 + t];
    h2[t] = s + b2[t]; }
  __syncthreads();
  if (t == 0) {
    float m = 0.f; for (int k = 0; k < 128; ++k) m += h2[k]; m *= (1.f / 128.f);
    float v = 0.f; for (int k = 0; k < 128; ++k) { float d = h2[k] - m; v += d * d; }
    red[0] = m; red[1] = rsqrtf(v * (1.f / 128.f) + 1e-5f);
  }
  __syncthreads();
  if (t < 128) e_init[t] = (h2[t] - red[0]) * red[1] * ls[t] + lb[t];
}

// ---------------------------------------------------------------------------
// node encoder: one-hot lookup -> relu -> 256x256 WMMA -> 256x128 WMMA -> LN
// ---------------------------------------------------------------------------
__global__ void k_node_enc(const float* __restrict__ actions, const float* W0,
                           const float* b0, const _Float16* W1s, const float* b1,
                           const _Float16* W2s, const float* b2, const float* ls,
                           const float* lb, float* __restrict__ x) {
  extern __shared__ char smem[];
  _Float16* lH0 = (_Float16*)smem;                      // 64*256
  _Float16* lH1 = lH0 + 64 * 256;                       // 64*256
  float*    lF  = (float*)(lH1 + 64 * 256);             // 64*128 f32
  const int lane = threadIdx.x & 31, wave = threadIdx.x >> 5;
  const int n0 = blockIdx.x * 64;

  for (int r = wave; r < 64; r += 8) {
    const int ty = (int)actions[(size_t)(n0 + r) * 4 + 3];
    const float* wr = W0 + ty * 256;
#pragma unroll
    for (int t = 0; t < 4; ++t) {
      const int f = t * 64 + lane * 2;
      const float2 w = *(const float2*)(wr + f);
      const float2 bb = *(const float2*)(b0 + f);
      st2h(lH0 + r * 256 + f, fmaxf(w.x + bb.x, 0.f), fmaxf(w.y + bb.y, 0.f));
    }
  }
  __syncthreads();
  mlp_layer<256, 256, true,  OM_LDSH>(lH0, W1s, b1, lH1, nullptr, nullptr, 0, 0);
  mlp_layer<256, 128, false, OM_LDSF>(lH1, W2s, b2, nullptr, lF, nullptr, 0, 0);

  // ln_post over 128 features, write x
  for (int r = wave; r < 64; r += 8) {
    float2 vals[2]; float sum = 0.f, sq = 0.f;
#pragma unroll
    for (int t = 0; t < 2; ++t) {
      const float2 v = *(const float2*)(lF + r * 128 + t * 64 + lane * 2);
      vals[t] = v; sum += v.x + v.y; sq += v.x * v.x + v.y * v.y;
    }
    sum = wred(sum); sq = wred(sq);
    const float mean = sum * (1.f / 128.f);
    const float inv = rsqrtf(sq * (1.f / 128.f) - mean * mean + 1e-5f);
#pragma unroll
    for (int t = 0; t < 2; ++t) {
      const int f = t * 64 + lane * 2;
      const float2 s = *(const float2*)(ls + f);
      const float2 b = *(const float2*)(lb + f);
      float2 o; o.x = (vals[t].x - mean) * inv * s.x + b.x;
      o.y = (vals[t].y - mean) * inv * s.y + b.y;
      *(float2*)(x + (size_t)(n0 + r) * 128 + f) = o;
    }
  }
}

// ---------------------------------------------------------------------------
// pos_emb MLP [3->128->128->128], accumulated into x
// ---------------------------------------------------------------------------
__global__ void k_pos_add(const float* __restrict__ actions, const float* W0,
                          const float* b0, const _Float16* W1s, const float* b1,
                          const _Float16* W2s, const float* b2, float* x) {
  extern __shared__ char smem[];
  _Float16* lH0 = (_Float16*)smem;                      // 64*128
  _Float16* lH1 = lH0 + 64 * 128;
  const int lane = threadIdx.x & 31, wave = threadIdx.x >> 5;
  const int n0 = blockIdx.x * 64;
  for (int r = wave; r < 64; r += 8) {
    const float* a = actions + (size_t)(n0 + r) * 4;
    const float px = a[0], py = a[1], pz = a[2];
#pragma unroll
    for (int t = 0; t < 2; ++t) {
      const int f = t * 64 + lane * 2;
      float2 o;
      o.x = fmaxf(px * W0[f]     + py * W0[128 + f]     + pz * W0[256 + f]     + b0[f],     0.f);
      o.y = fmaxf(px * W0[f + 1] + py * W0[128 + f + 1] + pz * W0[256 + f + 1] + b0[f + 1], 0.f);
      st2h(lH0 + r * 128 + f, o.x, o.y);
    }
  }
  __syncthreads();
  mlp_layer<128, 128, true,  OM_LDSH>(lH0, W1s, b1, lH1, nullptr, nullptr, 0, 0);
  mlp_layer<128, 128, false, OM_GADD>(lH1, W2s, b2, nullptr, nullptr, x, 128, n0);
}

// ---------------------------------------------------------------------------
// dis_emb MLP [1->128->128->128] on analytic pair distance, added into e
// ---------------------------------------------------------------------------
__global__ void k_dis_add(const float* __restrict__ actions, const float* W0,
                          const float* b0, const _Float16* W1s, const float* b1,
                          const _Float16* W2s, const float* b2, float* e) {
  extern __shared__ char smem[];
  _Float16* lH0 = (_Float16*)smem;
  _Float16* lH1 = lH0 + 64 * 128;
  const int lane = threadIdx.x & 31, wave = threadIdx.x >> 5;
  const size_t e0 = (size_t)blockIdx.x * 64;
  for (int r = wave; r < 64; r += 8) {
    const size_t eid = e0 + r;
    const int g = (int)(eid / 992), k = (int)(eid % 992);
    const int m = k / 31, j = k % 31, n = j + (j >= m ? 1 : 0);
    const float* pa = actions + (size_t)(g * 32 + m) * 4;
    const float* pb = actions + (size_t)(g * 32 + n) * 4;
    const float dx = pa[0] - pb[0], dy = pa[1] - pb[1], dz = pa[2] - pb[2];
    const float len = sqrtf(dx * dx + dy * dy + dz * dz);
#pragma unroll
    for (int t = 0; t < 2; ++t) {
      const int f = t * 64 + lane * 2;
      st2h(lH0 + r * 128 + f, fmaxf(len * W0[f] + b0[f], 0.f),
           fmaxf(len * W0[f + 1] + b0[f + 1], 0.f));
    }
  }
  __syncthreads();
  mlp_layer<128, 128, true,  OM_LDSH>(lH0, W1s, b1, lH1, nullptr, nullptr, 0, 0);
  mlp_layer<128, 128, false, OM_GADD>(lH1, W2s, b2, nullptr, nullptr, e, 128, (int)e0);
}

// ---------------------------------------------------------------------------
// per-edge GN MLP: ln_pre(concat(x[row],x[col],e,u[g])) -> 512->256->256->128
// ---------------------------------------------------------------------------
__global__ void k_edge_mlp(const float* __restrict__ x, const float* __restrict__ e,
                           const float* __restrict__ u, const float* lns,
                           const float* lnb, const _Float16* W0s, const float* b0,
                           const _Float16* W1s, const float* b1,
                           const _Float16* W2s, const float* b2,
                           float* __restrict__ e_new) {
  extern __shared__ char smem[];
  _Float16* lA  = (_Float16*)smem;                      // 64*512
  _Float16* lH1 = lA + 64 * 512;                        // 64*256
  _Float16* lH2 = lH1 + 64 * 256;                       // 64*256
  const int lane = threadIdx.x & 31, wave = threadIdx.x >> 5;
  const size_t e0 = (size_t)blockIdx.x * 64;

  for (int r = wave; r < 64; r += 8) {
    const size_t eid = e0 + r;
    const int g = (int)(eid / 992), k = (int)(eid % 992);
    const int m = k / 31, j = k % 31, n = j + (j >= m ? 1 : 0);
    const float* segs[4] = { x + (size_t)(g * 32 + m) * 128,
                             x + (size_t)(g * 32 + n) * 128,
                             e + eid * 128,
                             u + (size_t)g * 128 };
    float2 vals[8]; float sum = 0.f, sq = 0.f;
#pragma unroll
    for (int t = 0; t < 8; ++t) {
      const float2 v = *(const float2*)(segs[t >> 1] + (t & 1) * 64 + lane * 2);
      vals[t] = v; sum += v.x + v.y; sq += v.x * v.x + v.y * v.y;
    }
    sum = wred(sum); sq = wred(sq);
    const float mean = sum * (1.f / 512.f);
    const float inv = rsqrtf(sq * (1.f / 512.f) - mean * mean + 1e-5f);
#pragma unroll
    for (int t = 0; t < 8; ++t) {
      const int f = t * 64 + lane * 2;
      const float2 s = *(const float2*)(lns + f);
      const float2 b = *(const float2*)(lnb + f);
      st2h(lA + r * 512 + f, (vals[t].x - mean) * inv * s.x + b.x,
           (vals[t].y - mean) * inv * s.y + b.y);
    }
  }
  __syncthreads();
  mlp_layer<512, 256, true,  OM_LDSH>(lA,  W0s, b0, lH1, nullptr, nullptr, 0, 0);
  mlp_layer<256, 256, true,  OM_LDSH>(lH1, W1s, b1, lH2, nullptr, nullptr, 0, 0);
  mlp_layer<256, 128, false, OM_GST >(lH2, W2s, b2, nullptr, nullptr, e_new, 128, (int)e0);
}

// ---------------------------------------------------------------------------
// per-node GN MLP: ln_pre(concat(x,agg_r,agg_s,u[g])) -> 512->256->256->128
// ---------------------------------------------------------------------------
__global__ void k_node_mlp(const float* __restrict__ x, const float* __restrict__ ar,
                           const float* __restrict__ as, const float* __restrict__ u,
                           const float* lns, const float* lnb, const _Float16* W0s,
                           const float* b0, const _Float16* W1s, const float* b1,
                           const _Float16* W2s, const float* b2,
                           float* __restrict__ x_new) {
  extern __shared__ char smem[];
  _Float16* lA  = (_Float16*)smem;
  _Float16* lH1 = lA + 64 * 512;
  _Float16* lH2 = lH1 + 64 * 256;
  const int lane = threadIdx.x & 31, wave = threadIdx.x >> 5;
  const int n0 = blockIdx.x * 64;

  for (int r = wave; r < 64; r += 8) {
    const int nid = n0 + r, g = nid >> 5;
    const float* segs[4] = { x + (size_t)nid * 128, ar + (size_t)nid * 128,
                             as + (size_t)nid * 128, u + (size_t)g * 128 };
    float2 vals[8]; float sum = 0.f, sq = 0.f;
#pragma unroll
    for (int t = 0; t < 8; ++t) {
      const float2 v = *(const float2*)(segs[t >> 1] + (t & 1) * 64 + lane * 2);
      vals[t] = v; sum += v.x + v.y; sq += v.x * v.x + v.y * v.y;
    }
    sum = wred(sum); sq = wred(sq);
    const float mean = sum * (1.f / 512.f);
    const float inv = rsqrtf(sq * (1.f / 512.f) - mean * mean + 1e-5f);
#pragma unroll
    for (int t = 0; t < 8; ++t) {
      const int f = t * 64 + lane * 2;
      const float2 s = *(const float2*)(lns + f);
      const float2 b = *(const float2*)(lnb + f);
      st2h(lA + r * 512 + f, (vals[t].x - mean) * inv * s.x + b.x,
           (vals[t].y - mean) * inv * s.y + b.y);
    }
  }
  __syncthreads();
  mlp_layer<512, 256, true,  OM_LDSH>(lA,  W0s, b0, lH1, nullptr, nullptr, 0, 0);
  mlp_layer<256, 256, true,  OM_LDSH>(lH1, W1s, b1, lH2, nullptr, nullptr, 0, 0);
  mlp_layer<256, 128, false, OM_GST >(lH2, W2s, b2, nullptr, nullptr, x_new, 128, n0);
}

// ---------------------------------------------------------------------------
// global GN MLP: ln_pre(concat(u,node_agg,edge_agg)) -> 384->256->256->128,
// residual-fused: u += out
// ---------------------------------------------------------------------------
__global__ void k_glob_mlp(float* __restrict__ u, const float* __restrict__ na,
                           const float* __restrict__ ea, const float* lns,
                           const float* lnb, const _Float16* W0s, const float* b0,
                           const _Float16* W1s, const float* b1,
                           const _Float16* W2s, const float* b2) {
  extern __shared__ char smem[];
  _Float16* lA  = (_Float16*)smem;                      // 64*384
  _Float16* lH1 = lA + 64 * 384;
  _Float16* lH2 = lH1 + 64 * 256;
  const int lane = threadIdx.x & 31, wave = threadIdx.x >> 5;
  const int g0 = blockIdx.x * 64;

  for (int r = wave; r < 64; r += 8) {
    const int g = g0 + r;
    const float* segs[3] = { u + (size_t)g * 128, na + (size_t)g * 128,
                             ea + (size_t)g * 128 };
    float2 vals[6]; float sum = 0.f, sq = 0.f;
#pragma unroll
    for (int t = 0; t < 6; ++t) {
      const float2 v = *(const float2*)(segs[t >> 1] + (t & 1) * 64 + lane * 2);
      vals[t] = v; sum += v.x + v.y; sq += v.x * v.x + v.y * v.y;
    }
    sum = wred(sum); sq = wred(sq);
    const float mean = sum * (1.f / 384.f);
    const float inv = rsqrtf(sq * (1.f / 384.f) - mean * mean + 1e-5f);
#pragma unroll
    for (int t = 0; t < 6; ++t) {
      const int f = t * 64 + lane * 2;
      const float2 s = *(const float2*)(lns + f);
      const float2 b = *(const float2*)(lnb + f);
      st2h(lA + r * 384 + f, (vals[t].x - mean) * inv * s.x + b.x,
           (vals[t].y - mean) * inv * s.y + b.y);
    }
  }
  __syncthreads();
  mlp_layer<384, 256, true,  OM_LDSH>(lA,  W0s, b0, lH1, nullptr, nullptr, 0, 0);
  mlp_layer<256, 256, true,  OM_LDSH>(lH1, W1s, b1, lH2, nullptr, nullptr, 0, 0);
  mlp_layer<256, 128, false, OM_GADD>(lH2, W2s, b2, nullptr, nullptr, u, 128, g0);
}

// ---------------------------------------------------------------------------
// segment means (analytic edge enumeration, no atomics)
// ---------------------------------------------------------------------------
__global__ void k_node_agg(const float* __restrict__ e_new, float* __restrict__ ar,
                           float* __restrict__ as) {
  const int nid = blockIdx.x, g = nid >> 5, n = nid & 31, f = threadIdx.x;
  float sr = 0.f, ss = 0.f;
  for (int m = 0; m < 32; ++m) {
    if (m == n) continue;
    const size_t er = (size_t)g * 992 + m * 31 + (n - (n > m ? 1 : 0)); // col==n
    sr += e_new[er * 128 + f];
  }
  for (int j = 0; j < 31; ++j)                                          // row==n
    ss += e_new[((size_t)g * 992 + n * 31 + j) * 128 + f];
  ar[(size_t)nid * 128 + f] = sr * (1.f / 31.f);
  as[(size_t)nid * 128 + f] = ss * (1.f / 31.f);
}

__global__ void k_graph_agg(const float* __restrict__ src, float* __restrict__ dst,
                            int rows, float inv) {
  const int g = blockIdx.x, f = threadIdx.x;
  float s = 0.f;
  for (int r = 0; r < rows; ++r) s += src[((size_t)g * rows + r) * 128 + f];
  dst[(size_t)g * 128 + f] = s * inv;
}

// ---------------------------------------------------------------------------
// decoder: u[128,128] -> 256 -> 256 -> 1
// ---------------------------------------------------------------------------
__global__ void k_dec(const float* __restrict__ u, const _Float16* W0s,
                      const float* b0, const _Float16* W1s, const float* b1,
                      const float* W2, const float* b2, float* __restrict__ out) {
  extern __shared__ char smem[];
  _Float16* lA  = (_Float16*)smem;                      // 64*128
  _Float16* lH0 = lA + 64 * 128;                        // 64*256
  _Float16* lH1 = lH0 + 64 * 256;                       // 64*256
  const int lane = threadIdx.x & 31, wave = threadIdx.x >> 5;
  const int g0 = blockIdx.x * 64;
  for (int r = wave; r < 64; r += 8)
#pragma unroll
    for (int t = 0; t < 2; ++t) {
      const int f = t * 64 + lane * 2;
      const float2 v = *(const float2*)(u + (size_t)(g0 + r) * 128 + f);
      st2h(lA + r * 128 + f, v.x, v.y);
    }
  __syncthreads();
  mlp_layer<128, 256, true, OM_LDSH>(lA,  W0s, b0, lH0, nullptr, nullptr, 0, 0);
  mlp_layer<256, 256, true, OM_LDSH>(lH0, W1s, b1, lH1, nullptr, nullptr, 0, 0);
  for (int r = wave; r < 64; r += 8) {
    const v8h hv = *(const v8h*)(lH1 + r * 256 + lane * 8);
    float s = 0.f;
#pragma unroll
    for (int j = 0; j < 8; ++j) s += (float)hv[j] * W2[lane * 8 + j];
    s = wred(s);
    if (lane == 0) out[g0 + r] = s + b2[0];
  }
}

// ---------------------------------------------------------------------------
// host driver
// ---------------------------------------------------------------------------
extern "C" void kernel_launch(void* const* d_in, const int* in_sizes, int n_in,
                              void* d_out, int out_size, void* d_ws, size_t ws_size,
                              hipStream_t stream) {
  (void)in_sizes; (void)n_in; (void)out_size; (void)ws_size;
  constexpr int Gg = 128, NA = 32, Nn = Gg * NA;        // 4096 nodes
  constexpr size_t Ee = (size_t)Gg * NA * (NA - 1);     // 126976 edges

  int idx = 0;
  auto in = [&](void) { return (const float*)d_in[idx++]; };

  const float* actions = in();
  const float *neW0 = in(), *neW1 = in(), *neW2 = in(), *neb0 = in(), *neb1 = in(),
              *neb2 = in(), *nels = in(), *nelb = in();
  const float *eeW0 = in(), *eeW1 = in(), *eeW2 = in(), *eeb0 = in(), *eeb1 = in(),
              *eeb2 = in(), *eels = in(), *eelb = in();
  const float *peW0 = in(), *peW1 = in(), *peW2 = in(), *peb0 = in(), *peb1 = in(),
              *peb2 = in();
  const float *deW0 = in(), *deW1 = in(), *deW2 = in(), *deb0 = in(), *deb1 = in(),
              *deb2 = in();
  struct GNP { const float *W0, *W1, *W2, *b0, *b1, *b2, *ls, *lb; };
  GNP gn[3][3];
  for (int s = 0; s < 3; ++s)
    for (int p = 0; p < 3; ++p) {
      GNP& q = gn[s][p];
      q.W0 = in(); q.W1 = in(); q.W2 = in();
      q.b0 = in(); q.b1 = in(); q.b2 = in();
      q.ls = in(); q.lb = in();
    }
  const float *dcW0 = in(), *dcW1 = in(), *dcW2 = in(), *dcb0 = in(), *dcb1 = in(),
              *dcb2 = in();

  // ---- workspace carve-out -------------------------------------------------
  char* cur = (char*)d_ws;
  auto au = [](size_t v) { return (v + 255) & ~(size_t)255; };
  auto allocH = [&](size_t n) { _Float16* p = (_Float16*)cur; cur += au(n * 2); return p; };
  auto allocF = [&](size_t n) { float* p = (float*)cur; cur += au(n * 4); return p; };
  auto swz = [&](const float* s, int K, int N) {
    _Float16* d = allocH((size_t)K * N);
    k_wswz<<<dim3((unsigned)(((size_t)K * N + 255) / 256)), dim3(256), 0, stream>>>(
        s, d, K, N);
    return d;
  };

  // fragment-swizzled f16 weights (L2-resident across all tiles)
  _Float16* neW1s = swz(neW1, 256, 256);
  _Float16* neW2s = swz(neW2, 256, 128);
  _Float16* peW1s = swz(peW1, 128, 128);
  _Float16* peW2s = swz(peW2, 128, 128);
  _Float16* deW1s = swz(deW1, 128, 128);
  _Float16* deW2s = swz(deW2, 128, 128);
  struct GNH { _Float16 *W0, *W1, *W2; };
  GNH gnh[3][3];
  for (int s = 0; s < 3; ++s)
    for (int p = 0; p < 3; ++p) {
      const int k0 = (p == 2) ? 384 : 512;
      gnh[s][p].W0 = swz(gn[s][p].W0, k0, 256);
      gnh[s][p].W1 = swz(gn[s][p].W1, 256, 256);
      gnh[s][p].W2 = swz(gn[s][p].W2, 256, 128);
    }
  _Float16* dcW0s = swz(dcW0, 128, 256);
  _Float16* dcW1s = swz(dcW1, 256, 256);

  // f32 activation buffers
  float* x      = allocF((size_t)Nn * 128);
  float* x_new  = allocF((size_t)Nn * 128);
  float* agg_r  = allocF((size_t)Nn * 128);
  float* agg_s  = allocF((size_t)Nn * 128);
  float* u      = allocF((size_t)Gg * 128);
  float* nagg   = allocF((size_t)Gg * 128);
  float* eagg   = allocF((size_t)Gg * 128);
  float* e_init = allocF(128);
  float* e      = allocF(Ee * 128);
  float* e_new  = allocF(Ee * 128);

  const size_t EF = Ee * 128;
  const unsigned gEF = (unsigned)((EF + 255) / 256);
  const unsigned gEt = (unsigned)(Ee / 64);   // 1984 edge tiles
  const unsigned gNt = Nn / 64;               // 64 node tiles

  // LDS sizes (dynamic shared)
  const size_t L_BIG = 64 * 512 * 2 + 2 * 64 * 256 * 2;   // 131072
  const size_t L_GLB = 64 * 384 * 2 + 2 * 64 * 256 * 2;   // 114688
  const size_t L_ENC = 2 * 64 * 256 * 2 + 64 * 128 * 4;   //  98304
  const size_t L_EMB = 2 * 64 * 128 * 2;                  //  32768
  const size_t L_DEC = 64 * 128 * 2 + 2 * 64 * 256 * 2;   //  81920

  // ---- forward -------------------------------------------------------------
  k_zero<<<dim3(64), dim3(256), 0, stream>>>(u, Gg * 128);
  k_edge_enc<<<dim3(1), dim3(256), 0, stream>>>(eeW0, eeW1, eeW2, eeb0, eeb1,
                                                eeb2, eels, eelb, e_init);
  k_bcast_e<<<dim3(gEF), dim3(256), 0, stream>>>(e_init, e, EF);
  k_node_enc<<<dim3(gNt), dim3(256), L_ENC, stream>>>(
      actions, neW0, neb0, neW1s, neb1, neW2s, neb2, nels, nelb, x);

  for (int s = 0; s < 3; ++s) {
    k_pos_add<<<dim3(gNt), dim3(256), L_EMB, stream>>>(
        actions, peW0, peb0, peW1s, peb1, peW2s, peb2, x);
    k_dis_add<<<dim3(gEt), dim3(256), L_EMB, stream>>>(
        actions, deW0, deb0, deW1s, deb1, deW2s, deb2, e);

    const GNP &pe_ = gn[s][0], &pn_ = gn[s][1], &pg_ = gn[s][2];
    const GNH &he = gnh[s][0], &hn = gnh[s][1], &hg = gnh[s][2];

    k_edge_mlp<<<dim3(gEt), dim3(256), L_BIG, stream>>>(
        x, e, u, pe_.ls, pe_.lb, he.W0, pe_.b0, he.W1, pe_.b1, he.W2, pe_.b2, e_new);
    k_node_agg<<<dim3(Nn), dim3(128), 0, stream>>>(e_new, agg_r, agg_s);
    k_graph_agg<<<dim3(Gg), dim3(128), 0, stream>>>(e_new, eagg, NA * (NA - 1),
                                                    1.f / (NA * (NA - 1)));
    k_node_mlp<<<dim3(gNt), dim3(256), L_BIG, stream>>>(
        x, agg_r, agg_s, u, pn_.ls, pn_.lb, hn.W0, pn_.b0, hn.W1, pn_.b1, hn.W2,
        pn_.b2, x_new);
    k_graph_agg<<<dim3(Gg), dim3(128), 0, stream>>>(x_new, nagg, NA, 1.f / NA);
    k_glob_mlp<<<dim3(2), dim3(256), L_GLB, stream>>>(
        u, nagg, eagg, pg_.ls, pg_.lb, hg.W0, pg_.b0, hg.W1, pg_.b1, hg.W2, pg_.b2);

    k_axpy<<<dim3((unsigned)((size_t)Nn * 128 / 256)), dim3(256), 0, stream>>>(
        x, x_new, (size_t)Nn * 128);
    k_axpy<<<dim3(gEF), dim3(256), 0, stream>>>(e, e_new, EF);
  }

  k_dec<<<dim3(2), dim3(256), L_DEC, stream>>>(u, dcW0s, dcb0, dcW1s, dcb1, dcW2,
                                               dcb2, (float*)d_out);
}
